// AuxLossFreeGate_68882685493798
// MI455X (gfx1250) — matrix operand order repaired and checked by reading, not compiled, not run on hardware
//
#include <hip/hip_runtime.h>
#include <hip/hip_bf16.h>
#include <math.h>

// ---- gate hyperparameters ----
#define DIM        2048
#define NEXP       256
#define TOPK       8
#define NGROUPS    8
#define EPG        32      // experts per group
#define TOPKG      4       // top groups kept
#define ROUTE_SCALE 2.5f
#define TTOK       32768

// ---- tiling ----
#define MTILE   64                 // tokens per block
#define BLOCK   256                // 8 waves (wave32)
#define KC      256                // K chunk staged in LDS
#define A_STRIDE (KC + 8)          // bf16 elems, padded (528B row -> bank-conflict free b128)
#define S_STRIDE 260               // f32 score row stride (260 % 64 == 4)

typedef __bf16 bf16_t;
typedef __attribute__((ext_vector_type(16))) __bf16 v16bf;
typedef __attribute__((ext_vector_type(8)))  __bf16 v8bf;
typedef __attribute__((ext_vector_type(4)))  __bf16 v4bf;
typedef __attribute__((ext_vector_type(8)))  float  v8f;

// One-time weight f32 -> bf16 conversion into workspace (2MB -> 1MB, L2-resident).
__global__ void AuxGate_wcvt(const float* __restrict__ w, bf16_t* __restrict__ wb) {
    int i = (blockIdx.x * blockDim.x + threadIdx.x) * 4;
    float4 v = *(const float4*)(w + i);
    v4bf o;
    o[0] = (bf16_t)v.x; o[1] = (bf16_t)v.y; o[2] = (bf16_t)v.z; o[3] = (bf16_t)v.w;
    *(v4bf*)(wb + i) = o;
}

// Fused: scores = sigmoid(x @ W^T)  (bf16 WMMA)  +  group-limited top-k routing.
__global__ __launch_bounds__(BLOCK)
void AuxGate_main(const float* __restrict__ x,
                  const bf16_t* __restrict__ wb,
                  const float* __restrict__ bias,
                  float* __restrict__ wout,
                  int* __restrict__ iout) {
    __shared__ bf16_t ldsA[MTILE * A_STRIDE];        // staged x tile (bf16)
    __shared__ float  ldsS[MTILE * S_STRIDE];        // raw sigmoid scores
    __shared__ float  ldsBias[NEXP];

    const int tid  = threadIdx.x;
    const int wave = tid >> 5;        // 0..7
    const int lane = tid & 31;
    const int g    = lane >> 4;       // half-wave group (0/1)
    const int lm   = lane & 15;
    const int tok0 = blockIdx.x * MTILE;

    ldsBias[tid] = bias[tid];         // 256 threads == 256 experts

    // wave handles M-subtiles 0..3 and E-tiles (2*wave + t), t in {0,1}
    v8f acc[4][2];
    #pragma unroll
    for (int ms = 0; ms < 4; ++ms)
        #pragma unroll
        for (int t = 0; t < 2; ++t)
            acc[ms][t] = (v8f){};

    // staging layout: 4 threads per row, 64 f32 per thread
    const int ldrow  = tid >> 2;                 // 0..63
    const int ldcol0 = (tid & 3) * 4;            // float4 base

    for (int kc = 0; kc < DIM; kc += KC) {
        // ---- cooperative f32 -> bf16 staging of x[tok0..tok0+63][kc..kc+KC) ----
        const float* xrow = x + (size_t)(tok0 + ldrow) * DIM + kc;
        #pragma unroll
        for (int j = 0; j < KC / 16; ++j) {      // 16 iters of float4
            int col = ldcol0 + j * 16;
            float4 v = *(const float4*)(xrow + col);
            v4bf o;
            o[0] = (bf16_t)v.x; o[1] = (bf16_t)v.y;
            o[2] = (bf16_t)v.z; o[3] = (bf16_t)v.w;
            *(v4bf*)(ldsA + ldrow * A_STRIDE + col) = o;
        }
        __syncthreads();

        #pragma unroll
        for (int ks = 0; ks < KC; ks += 32) {
            // A fragments: lane holds M = ms*16 + lm; halves 0..7 -> K = g*8+i,
            // halves 8..15 -> K = 16+g*8+i  (ISA 16-bit A 16x32 layout)
            v16bf afrag[4];
            #pragma unroll
            for (int ms = 0; ms < 4; ++ms) {
                const bf16_t* ap = ldsA + (ms * 16 + lm) * A_STRIDE + ks + g * 8;
                v8bf alo = *(const v8bf*)(ap);
                v8bf ahi = *(const v8bf*)(ap + 16);
                afrag[ms] = __builtin_shufflevector(alo, ahi,
                    0, 1, 2, 3, 4, 5, 6, 7, 8, 9, 10, 11, 12, 13, 14, 15);
            }
            // B fragments: lane holds N = lm (expert), halves 0..7 -> K = g*8+i, 8..15 -> K = 16+g*8+i;
            // B[k][n] = weight[e0+n][k] -> two 16-byte runs (16 bytes apart) of a weight row
            v16bf bfrag[2];
            #pragma unroll
            for (int t = 0; t < 2; ++t) {
                int e0 = (wave * 2 + t) * 16;
                const bf16_t* bp = wb + (size_t)(e0 + lm) * DIM + kc + ks + g * 8;
                v8bf blo = *(const v8bf*)(bp);
                v8bf bhi = *(const v8bf*)(bp + 16);
                bfrag[t] = __builtin_shufflevector(blo, bhi,
                    0, 1, 2, 3, 4, 5, 6, 7, 8, 9, 10, 11, 12, 13, 14, 15);
            }
            #pragma unroll
            for (int ms = 0; ms < 4; ++ms)
                #pragma unroll
                for (int t = 0; t < 2; ++t)
                    acc[ms][t] = __builtin_amdgcn_wmma_f32_16x16x32_bf16(
                        false, afrag[ms], false, bfrag[t],
                        (short)0, acc[ms][t], false, false);
        }
        __syncthreads();
    }

    // ---- sigmoid + scatter scores to LDS (C/D layout: VGPR r -> M = r + 8*g) ----
    #pragma unroll
    for (int ms = 0; ms < 4; ++ms)
        #pragma unroll
        for (int t = 0; t < 2; ++t) {
            int e0 = (wave * 2 + t) * 16;
            #pragma unroll
            for (int r = 0; r < 8; ++r) {
                float a = acc[ms][t][r];
                float s = 1.0f / (1.0f + __expf(-a));
                ldsS[(ms * 16 + r + 8 * g) * S_STRIDE + e0 + lm] = s;
            }
        }
    __syncthreads();

    // ---- group-limited top-k routing: one thread per token ----
    if (tid < MTILE) {
        const float* srow = ldsS + tid * S_STRIDE;

        // group score = sum of top-2 biased scores in each group of 32
        float gs[NGROUPS];
        #pragma unroll
        for (int grp = 0; grp < NGROUPS; ++grp) {
            float m1 = -3.0e38f, m2 = -3.0e38f;
            for (int j = 0; j < EPG; ++j) {
                int e = grp * EPG + j;
                float bv = srow[e] + ldsBias[e];
                if (bv > m1) { m2 = m1; m1 = bv; }
                else if (bv > m2) { m2 = bv; }
            }
            gs[grp] = m1 + m2;
        }
        // top-4 groups
        unsigned keepMask = 0;
        for (int k = 0; k < TOPKG; ++k) {
            int best = 0; float bv = -3.0e38f;
            for (int grp = 0; grp < NGROUPS; ++grp)
                if (!((keepMask >> grp) & 1u) && gs[grp] > bv) { bv = gs[grp]; best = grp; }
            keepMask |= 1u << best;
        }
        // top-8 experts over biased scores within kept groups
        unsigned long long chosen[4] = {0ull, 0ull, 0ull, 0ull};
        float wts[TOPK]; int idxs[TOPK]; float wsum = 0.0f;
        for (int k = 0; k < TOPK; ++k) {
            int best = 0; float bv = -3.0e38f;
            for (int e = 0; e < NEXP; ++e) {
                if (!((keepMask >> (e >> 5)) & 1u)) continue;
                if ((chosen[e >> 6] >> (e & 63)) & 1ull) continue;
                float b = srow[e] + ldsBias[e];
                if (b > bv) { bv = b; best = e; }
            }
            chosen[best >> 6] |= 1ull << (best & 63);
            float rw = srow[best];     // weights come from RAW sigmoid scores
            wts[k] = rw; idxs[k] = best; wsum += rw;
        }
        float scale = ROUTE_SCALE / fmaxf(wsum, 1e-10f);
        int tok = tok0 + tid;
        #pragma unroll
        for (int k = 0; k < TOPK; ++k) {
            wout[(size_t)tok * TOPK + k] = wts[k] * scale;
            iout[(size_t)tok * TOPK + k] = idxs[k];
        }
    }
}

extern "C" void kernel_launch(void* const* d_in, const int* in_sizes, int n_in,
                              void* d_out, int out_size, void* d_ws, size_t ws_size,
                              hipStream_t stream) {
    const float* x    = (const float*)d_in[0];   // (T, D) f32
    const float* w    = (const float*)d_in[1];   // (E, D) f32
    const float* bias = (const float*)d_in[2];   // (E,)   f32

    float* wout = (float*)d_out;                           // (T, 8) f32
    int*   iout = (int*)d_out + (size_t)TTOK * TOPK;       // (T, 8) int32 bits

    bf16_t* wb = (bf16_t*)d_ws;                            // 1MB bf16 weight

    // 256*2048 elems / 4 per thread / 256 threads = 512 blocks
    AuxGate_wcvt<<<(NEXP * DIM) / (4 * 256), 256, 0, stream>>>(w, wb);
    AuxGate_main<<<TTOK / MTILE, BLOCK, 0, stream>>>(x, wb, bias, wout, iout);
}
